// GraphFingerprintsModelAttentionCNN_80573586473455
// MI455X (gfx1250) — matrix operand order repaired and checked by reading, not compiled, not run on hardware
//
#include <hip/hip_runtime.h>
#include <cstdint>

// ---------------------------------------------------------------------------
// GraphFingerprintsModelAttentionCNN forward for MI455X (gfx1250, wave32).
// Dense GEMMs -> v_wmma_f32_16x16x32_bf16 (bf16 in, fp32 acc).  LDS tiles are
// staged in *fragment order* (each WMMA operand = 2 x ds_load_b128) through a
// register-staged double-buffered pipeline: next tile's global_load_b128s are
// issued before the WMMA phase and only waited on by the ds_store into the
// alternate buffer; all fragments are preloaded so DS and WMMA overlap.
// ---------------------------------------------------------------------------

constexpr int Gd   = 256;
constexpr int NFd  = 39;
constexpr int EFd  = 10;
constexpr int SOLV = 1024;
constexpr int DSM  = 2192;
constexpr int Bsz  = 256;
constexpr int NPG  = 48;
constexpr int Vn   = Bsz * NPG;   // 12288 nodes
constexpr int En   = 4 * Vn;      // 49152 edges
constexpr int Td   = 2;

typedef __attribute__((ext_vector_type(16))) __bf16 v16bf;
typedef __attribute__((ext_vector_type(8)))  float  v8f;

__device__ __forceinline__ unsigned short f2bf(float f) {
  union { float f; unsigned u; } v; v.f = f;
  unsigned u = v.u;
  u += 0x7FFFu + ((u >> 16) & 1u);          // round-to-nearest-even
  return (unsigned short)(u >> 16);
}

__device__ __forceinline__ float apply_act(float x, int act) {
  if (act == 1) return x > 0.f ? x : 0.01f * x;          // leaky_relu
  if (act == 2) return x > 0.f ? x : 0.f;                // relu
  if (act == 3) return x > 0.f ? x : (__expf(x) - 1.f);  // elu
  return x;
}

// float atomic-max via ordered-int trick (init with -1e30f)
__device__ __forceinline__ void atomicMaxF(float* addr, float val) {
  if (val >= 0.f) atomicMax((int*)addr, __float_as_int(val));
  else            atomicMin((unsigned int*)addr, (unsigned int)__float_as_int(val));
}

// ---------------------------------------------------------------------------
// WMMA GEMM:  C[M,ldc] = act(A[M,K](bf16) @ W[N,K]^T(bf16) + bias)
// Block tile 64x128, 8 waves (4 in M x 2 in N); each wave: 16x64 = 4 accs.
// Requires: M%64==0, N%128==0, K%32==0 (true for every GEMM in this model).
// ---------------------------------------------------------------------------
#define ASTRIDE 40   // ushorts per A row slot (32 data + 8 pad)
#define BSTRIDE 40   // ushorts per B column slot

__global__ __launch_bounds__(256)
void gemm_bf16_wmma(const unsigned short* __restrict__ A,
                    const unsigned short* __restrict__ W,
                    const float* __restrict__ bias,
                    float* __restrict__ C,
                    unsigned short* __restrict__ Cbf,
                    int M, int N, int K, int ldc, int act)
{
  // A rows stored contiguous in k: As[m][k]
  __shared__ __align__(16) unsigned short As[2][64 * ASTRIDE];
  // B columns stored contiguous in k: Bs[n][k]
  __shared__ __align__(16) unsigned short Bs[2][128 * BSTRIDE];

  const int tid   = threadIdx.x;
  const int wave  = tid >> 5;
  const int lane  = tid & 31;
  const int bm    = blockIdx.x * 64;
  const int bn    = blockIdx.y * 128;
  const int wm    = (wave >> 1) * 16;        // 0,16,32,48
  const int wn    = (wave & 1) * 64;         // 0,64
  const int lmod  = lane & 15;
  const int lhalf = lane >> 4;               // 0 or 1
  const int permA[4] = {0, 2, 1, 3};         // unused; LDS slot = k-group (tg)

  v8f acc[4] = {{}, {}, {}, {}};

  // register-staged tile: each thread owns one 16B slice of A and two of B
  const int trow = tid >> 2;                 // 0..63
  const int tg   = tid & 3;                  // k-group (8 bf16 = 16 bytes)
  const unsigned short* Ap  = &A[(size_t)(bm + trow)      * K + tg * 8];
  const unsigned short* Wp0 = &W[(size_t)(bn + trow)      * K + tg * 8];
  const unsigned short* Wp1 = &W[(size_t)(bn + 64 + trow) * K + tg * 8];
  uint4 ra, rb0, rb1;

  auto gload = [&](int k0) {
    ra  = *(const uint4*)(Ap  + k0);
    rb0 = *(const uint4*)(Wp0 + k0);
    rb1 = *(const uint4*)(Wp1 + k0);
  };
  auto lstore = [&](int buf) {
    *(uint4*)&As[buf][trow * ASTRIDE + tg * 8]      = ra;
    *(uint4*)&Bs[buf][trow * BSTRIDE + tg * 8]             = rb0;
    *(uint4*)&Bs[buf][(64 + trow) * BSTRIDE + tg * 8]      = rb1;
  };

  gload(0);
  lstore(0);
  int cur = 0;

  union Frag { uint4 q[2]; v16bf v; };

  for (int k0 = 0; k0 < K; k0 += 32) {
    __syncthreads();                         // staged buffer 'cur' is ready
    const bool more = (k0 + 32) < K;
    if (more) gload(k0 + 32);                // issue next tile loads; they stay
                                             // outstanding through the WMMAs
    // ---- preload ALL fragments (10 x ds_load_b128), then run 4 WMMAs so the
    //      compiler can use partial s_wait_dscnt and overlap DS with WMMA.
    Frag fa, fb[4];
    {
      const unsigned short* p = &As[cur][(wm + lmod) * ASTRIDE + lhalf * 16];
      fa.q[0] = *(const uint4*)p;
      fa.q[1] = *(const uint4*)(p + 8);
    }
#pragma unroll
    for (int j = 0; j < 4; ++j) {
      const unsigned short* p = &Bs[cur][(wn + j * 16 + lmod) * BSTRIDE + lhalf * 16];
      fb[j].q[0] = *(const uint4*)p;
      fb[j].q[1] = *(const uint4*)(p + 8);
    }
#pragma unroll
    for (int j = 0; j < 4; ++j)
      acc[j] = __builtin_amdgcn_wmma_f32_16x16x32_bf16(false, fa.v, false, fb[j].v,
                                                       (short)0, acc[j], false, false);
    if (more) lstore(cur ^ 1);               // waits on the global loads here,
    cur ^= 1;                                // after a full compute phase
  }

  // C/D layout: reg r -> row wm + lhalf*8 + r, col = lmod within 16-col group
  const int rbase = bm + wm + lhalf * 8;
#pragma unroll
  for (int j = 0; j < 4; ++j) {
    int cj = bn + wn + j * 16 + lmod;
    float bj = bias ? bias[cj] : 0.f;
#pragma unroll
    for (int r = 0; r < 8; ++r) {
      float v = apply_act(acc[j][r] + bj, act);
      size_t o = (size_t)(rbase + r) * ldc + cj;
      if (C)   C[o] = v;
      if (Cbf) Cbf[o] = f2bf(v);
    }
  }
}

// ---------------------------------------------------------------------------
// Packing / cast kernels
// ---------------------------------------------------------------------------
__global__ void fill_f32(float* p, float v, int n) {
  int i = blockIdx.x * 256 + threadIdx.x; if (i < n) p[i] = v;
}
__global__ void cast_bf(const float* __restrict__ x, unsigned short* __restrict__ y, int n) {
  int i = blockIdx.x * 256 + threadIdx.x; if (i < n) y[i] = f2bf(x[i]);
}
__global__ void elu_cast_bf(const float* __restrict__ x, unsigned short* __restrict__ y, int n) {
  int i = blockIdx.x * 256 + threadIdx.x;
  if (i < n) { float v = x[i]; v = v > 0.f ? v : (__expf(v) - 1.f); y[i] = f2bf(v); }
}
__global__ void pack_pad(const float* __restrict__ s, unsigned short* __restrict__ d,
                         int rows, int sc, int dc) {
  int i = blockIdx.x * 256 + threadIdx.x;
  if (i >= rows * dc) return;
  int r = i / dc, c = i - r * dc;
  d[i] = f2bf(c < sc ? s[(size_t)r * sc + c] : 0.f);
}
// edge GEMM input: [nf[src] | edge_feats | 0-pad] -> [E,64] bf16
__global__ void pack_edge_in(const float* __restrict__ nf, const float* __restrict__ ef,
                             const int* __restrict__ srcI, unsigned short* __restrict__ d) {
  int e = blockIdx.x, c = threadIdx.x;   // blockDim 64
  float v = 0.f;
  if (c < NFd)            v = nf[(size_t)srcI[e] * NFd + c];
  else if (c < NFd + EFd) v = ef[(size_t)e * EFd + (c - NFd)];
  d[(size_t)e * 64 + c] = f2bf(v);
}
__global__ void pack_solvent(const float* __restrict__ fps, unsigned short* __restrict__ d, int n) {
  int i = blockIdx.x * 256 + threadIdx.x;
  if (i >= n) return;
  int b = i >> 10, c = i & 1023;
  d[i] = f2bf(fps[(size_t)b * (SOLV + DSM) + c]);
}
__global__ void copy_cols(const float* __restrict__ s, float* __restrict__ d, int cols, int ldd) {
  int b = blockIdx.x, c = threadIdx.x;
  d[(size_t)b * ldd + c] = s[(size_t)b * cols + c];
}

// ---------------------------------------------------------------------------
// lrelu( dot( [maybe-relu(X[xi]) , Y[yi]] , w[0..511] ) + b ) -> out, 1 wave/row
// ---------------------------------------------------------------------------
__global__ __launch_bounds__(256)
void concat_dot(const float* __restrict__ X, const int* __restrict__ xi,
                const float* __restrict__ Y, const int* __restrict__ yi,
                const float* __restrict__ w, const float* __restrict__ bias,
                float* __restrict__ out, int n, int relu_x)
{
  int row = blockIdx.x * 8 + (threadIdx.x >> 5);
  int lane = threadIdx.x & 31;
  if (row >= n) return;
  const float* xp = X + (size_t)(xi ? xi[row] : row) * Gd;
  const float* yp = Y + (size_t)(yi ? yi[row] : row) * Gd;
  float s = 0.f;
#pragma unroll 4
  for (int i = lane; i < Gd; i += 32) {
    float xv = xp[i]; if (relu_x) xv = fmaxf(xv, 0.f);
    s += xv * w[i] + yp[i] * w[Gd + i];
  }
#pragma unroll
  for (int off = 16; off >= 1; off >>= 1) s += __shfl_xor(s, off, 32);
  if (lane == 0) { float v = s + bias[0]; out[row] = v > 0.f ? v : 0.01f * v; }
}

// segment softmax pieces (V segments, atomic based)
__global__ void seg_max(const float* __restrict__ lg, const int* __restrict__ seg,
                        float* __restrict__ m, int n) {
  int i = blockIdx.x * 256 + threadIdx.x;
  if (i < n) atomicMaxF(&m[seg[i]], lg[i]);
}
__global__ void seg_expsum(const float* __restrict__ lg, const int* __restrict__ seg,
                           const float* __restrict__ m, float* __restrict__ ex,
                           float* __restrict__ ss, int n) {
  int i = blockIdx.x * 256 + threadIdx.x;
  if (i >= n) return;
  float e = __expf(lg[i] - m[seg[i]]);
  ex[i] = e;
  atomicAdd(&ss[seg[i]], e);
}
// ctx[dst[e]] += (ex/ss)[e] * val[e]        (one block per edge, 256 chans)
__global__ void scatter_msg(const float* __restrict__ val, const float* __restrict__ ex,
                            const float* __restrict__ ss, const int* __restrict__ dstI,
                            float* __restrict__ o) {
  int e = blockIdx.x, c = threadIdx.x;
  int d = dstI[e];
  float a = ex[e] / ss[d];
  atomicAdd(&o[(size_t)d * Gd + c], a * val[(size_t)e * Gd + c]);
}
// ctx2[dst[e]] += a2[e] * hv[src[e]]
__global__ void scatter_src(const float* __restrict__ hv, const float* __restrict__ ex,
                            const float* __restrict__ ss, const int* __restrict__ srcI,
                            const int* __restrict__ dstI, float* __restrict__ o) {
  int e = blockIdx.x, c = threadIdx.x;
  int d = dstI[e];
  float a = ex[e] / ss[d];
  atomicAdd(&o[(size_t)d * Gd + c], a * hv[(size_t)srcI[e] * Gd + c]);
}

// GRU combine: r=sig(ir+hr) z=sig(iz+hz) n=tanh(in + r*hn); out=(1-z)n+z*h
__global__ void gru_combine(const float* __restrict__ gi, const float* __restrict__ gh,
                            const float* __restrict__ h, float* __restrict__ ho,
                            unsigned short* __restrict__ hob, int M, int do_relu) {
  int t = blockIdx.x * 256 + threadIdx.x;
  if (t >= M * Gd) return;
  int m = t >> 8, c = t & 255;
  const float* gim = gi + (size_t)m * 3 * Gd;
  const float* ghm = gh + (size_t)m * 3 * Gd;
  float r = 1.f / (1.f + __expf(-(gim[c] + ghm[c])));
  float z = 1.f / (1.f + __expf(-(gim[Gd + c] + ghm[Gd + c])));
  float nn = tanhf(gim[2 * Gd + c] + r * ghm[2 * Gd + c]);
  float v = (1.f - z) * nn + z * h[t];
  if (do_relu) v = fmaxf(v, 0.f);
  if (ho)  ho[t] = v;
  if (hob) hob[t] = f2bf(v);
}

// per-graph softmax over NPG contiguous nodes (block of 64 per graph)
__global__ void graph_softmax(const float* __restrict__ z, float* __restrict__ an) {
  __shared__ float red[64];
  int g = blockIdx.x, t = threadIdx.x;
  float v = (t < NPG) ? z[g * NPG + t] : -1e30f;
  red[t] = v; __syncthreads();
  for (int s = 32; s >= 1; s >>= 1) { if (t < s) red[t] = fmaxf(red[t], red[t + s]); __syncthreads(); }
  float mx = red[0]; __syncthreads();
  float e = (t < NPG) ? __expf(v - mx) : 0.f;
  red[t] = e; __syncthreads();
  for (int s = 32; s >= 1; s >>= 1) { if (t < s) red[t] += red[t + s]; __syncthreads(); }
  if (t < NPG) an[g * NPG + t] = e / red[0];
}
// g[g][c] = sum_n (an? an[n]:1) * h[n][c]   (block per graph, 256 threads)
__global__ void graph_pool(const float* __restrict__ h, const float* __restrict__ an,
                           float* __restrict__ g) {
  int gg = blockIdx.x, c = threadIdx.x;
  float s = 0.f;
#pragma unroll 4
  for (int n = 0; n < NPG; ++n) {
    float w = an ? an[gg * NPG + n] : 1.f;
    s += w * h[(size_t)(gg * NPG + n) * Gd + c];
  }
  g[(size_t)gg * Gd + c] = s;
}

// fused smiles conv (k=3, pad 1) + position softmax + attn-pool + max-pool
__global__ __launch_bounds__(256)
void smiles_conv(const float* __restrict__ fps,
                 const float* __restrict__ Wcf, const float* __restrict__ bcf,
                 const float* __restrict__ Wca, const float* __restrict__ bca,
                 float* __restrict__ comb, int ldc)
{
  __shared__ float xs[DSM + 2];
  int b = blockIdx.x;
  const float* x = fps + (size_t)b * (SOLV + DSM) + SOLV;
  for (int i = threadIdx.x; i < DSM + 2; i += 256)
    xs[i] = (i == 0 || i == DSM + 1) ? 0.f : x[i - 1];
  __syncthreads();
  int wave = threadIdx.x >> 5, lane = threadIdx.x & 31;
  int c = blockIdx.y * 8 + wave;
  float wf0 = Wcf[c * 3], wf1 = Wcf[c * 3 + 1], wf2 = Wcf[c * 3 + 2], fb = bcf[c];
  float wa0 = Wca[c * 3], wa1 = Wca[c * 3 + 1], wa2 = Wca[c * 3 + 2], ab = bca[c];
  float amax = -1e30f;
  for (int p = lane; p < DSM; p += 32)
    amax = fmaxf(amax, wa0 * xs[p] + wa1 * xs[p + 1] + wa2 * xs[p + 2] + ab);
#pragma unroll
  for (int off = 16; off >= 1; off >>= 1) amax = fmaxf(amax, __shfl_xor(amax, off, 32));
  float esum = 0.f, num = 0.f, fmx = -1e30f;
  for (int p = lane; p < DSM; p += 32) {
    float fv = wf0 * xs[p] + wf1 * xs[p + 1] + wf2 * xs[p + 2] + fb;
    float av = wa0 * xs[p] + wa1 * xs[p + 1] + wa2 * xs[p + 2] + ab;
    float e = __expf(av - amax);
    esum += e; num += fv * e; fmx = fmaxf(fmx, fv);
  }
#pragma unroll
  for (int off = 16; off >= 1; off >>= 1) {
    esum += __shfl_xor(esum, off, 32);
    num  += __shfl_xor(num,  off, 32);
    fmx  = fmaxf(fmx, __shfl_xor(fmx, off, 32));
  }
  if (lane == 0) {
    comb[(size_t)b * ldc + 2 * Gd + c] = num / esum;   // attn_out
    comb[(size_t)b * ldc + 3 * Gd + c] = fmx;          // pooled
  }
}

// out[b] = dot(hid[b], Wp2) + bp2      (1 wave / row, K=128)
__global__ void final_dot(const float* __restrict__ hid, const float* __restrict__ w,
                          const float* __restrict__ b, float* __restrict__ out) {
  int row = blockIdx.x * 8 + (threadIdx.x >> 5);
  int lane = threadIdx.x & 31;
  float s = 0.f;
#pragma unroll
  for (int i = lane; i < 128; i += 32) s += hid[(size_t)row * 128 + i] * w[i];
#pragma unroll
  for (int off = 16; off >= 1; off >>= 1) s += __shfl_xor(s, off, 32);
  if (lane == 0) out[row] = s + b[0];
}

// ---------------------------------------------------------------------------
extern "C" void kernel_launch(void* const* d_in, const int* in_sizes, int n_in,
                              void* d_out, int out_size, void* d_ws, size_t ws_size,
                              hipStream_t stream)
{
  (void)in_sizes; (void)n_in; (void)out_size; (void)ws_size;
#define IN_F(i) ((const float*)d_in[(i)])
#define IN_I(i) ((const int*)d_in[(i)])
  const float* node_feats = IN_F(0);
  const float* edge_feats = IN_F(1);
  const float* fps        = IN_F(2);
  const int*   srcI       = IN_I(3);
  const int*   dstI       = IN_I(4);
  const int*   n2g        = IN_I(5);
  // params (setup_inputs insertion order)
  const float *W_pn = IN_F(6),  *b_pn = IN_F(7);
  const float *W_pe1 = IN_F(8), *b_pe1 = IN_F(9);
  const float *W_pe2 = IN_F(10),*b_pe2 = IN_F(11);
  const float *W_et = IN_F(12), *b_et = IN_F(13);
  const float *Wih1 = IN_F(14), *bih1 = IN_F(15);
  const float *Whh1 = IN_F(16), *bhh1 = IN_F(17);
  const float *W_pe = IN_F(18), *b_pe = IN_F(19);
  const float *W_pnode = IN_F(20), *b_pnode = IN_F(21);
  const float *Wih2 = IN_F(22), *bih2 = IN_F(23);
  const float *Whh2 = IN_F(24), *bhh2 = IN_F(25);
  const float *W_logit = IN_F(26), *b_logit = IN_F(27);
  const float *W_proj = IN_F(28),  *b_proj = IN_F(29);
  const float *Wih_r = IN_F(30), *bih_r = IN_F(31);
  const float *Whh_r = IN_F(32), *bhh_r = IN_F(33);
  const float *Wcf = IN_F(34), *bcf = IN_F(35);
  const float *Wca = IN_F(36), *bca = IN_F(37);
  const float *Ws1 = IN_F(38), *bs1 = IN_F(39);
  const float *Ws2 = IN_F(40), *bs2 = IN_F(41);
  const float *Wp1 = IN_F(42), *bp1 = IN_F(43);
  const float *Wp2 = IN_F(44), *bp2 = IN_F(45);

  char* base = (char*)d_ws;
  size_t off = 0;
  auto alloc = [&](size_t bytes) -> void* {
    void* p = base + off; off += (bytes + 255) & ~(size_t)255; return p;
  };
  // bf16 weights
  auto* WpnB    = (unsigned short*)alloc((size_t)Gd * 64 * 2);
  auto* Wpe1B   = (unsigned short*)alloc((size_t)Gd * 64 * 2);
  auto* WetB    = (unsigned short*)alloc((size_t)Gd * Gd * 2);
  auto* Wih1B   = (unsigned short*)alloc((size_t)3 * Gd * Gd * 2);
  auto* Whh1B   = (unsigned short*)alloc((size_t)3 * Gd * Gd * 2);
  auto* WpnodeB = (unsigned short*)alloc((size_t)Gd * Gd * 2);
  auto* Wih2B   = (unsigned short*)alloc((size_t)3 * Gd * Gd * 2);
  auto* Whh2B   = (unsigned short*)alloc((size_t)3 * Gd * Gd * 2);
  auto* WprojB  = (unsigned short*)alloc((size_t)Td * Gd * Gd * 2);
  auto* WihrB   = (unsigned short*)alloc((size_t)Td * 3 * Gd * Gd * 2);
  auto* WhhrB   = (unsigned short*)alloc((size_t)Td * 3 * Gd * Gd * 2);
  auto* Ws1B    = (unsigned short*)alloc((size_t)Gd * SOLV * 2);
  auto* Ws2B    = (unsigned short*)alloc((size_t)Gd * Gd * 2);
  auto* Wp1B    = (unsigned short*)alloc((size_t)128 * SOLV * 2);
  // activations  (big regions aliased across phases)
  auto* XnB   = (unsigned short*)alloc((size_t)Vn * 64 * 2);
  auto* EinB  = (unsigned short*)alloc((size_t)En * 64 * 2);
  float* hvF  = (float*)alloc((size_t)Vn * Gd * 4);          // hv_new, later h2
  auto* hvB   = (unsigned short*)alloc((size_t)Vn * Gd * 2);
  float* R1   = (float*)alloc((size_t)En * Gd * 4);          // he1_f32 -> gi
  float* R2   = (float*)alloc((size_t)En * Gd * 4);          // et_out  -> gh
  auto* he1B  = (unsigned short*)alloc((size_t)En * Gd * 2); // later hvp_f32
  float* lgBuf = (float*)alloc((size_t)En * 4);              // logits / nodeZ
  float* exBuf = (float*)alloc((size_t)En * 4);              // exp / an
  float* mBuf  = (float*)alloc((size_t)Vn * 4);
  float* sBuf  = (float*)alloc((size_t)Vn * 4);
  float* ctxF  = (float*)alloc((size_t)Vn * Gd * 4);
  auto* ctxB   = (unsigned short*)alloc((size_t)Vn * Gd * 2);
  float* hF    = (float*)alloc((size_t)Vn * Gd * 4);
  auto* hB     = (unsigned short*)alloc((size_t)Vn * Gd * 2);
  float* gfeatF = (float*)alloc((size_t)Bsz * Gd * 4);
  auto* gfeatB  = (unsigned short*)alloc((size_t)Bsz * Gd * 2);
  float* grprF  = (float*)alloc((size_t)Bsz * Gd * 4);
  auto* grprB   = (unsigned short*)alloc((size_t)Bsz * Gd * 2);
  float* giR = (float*)alloc((size_t)Bsz * 3 * Gd * 4);
  float* ghR = (float*)alloc((size_t)Bsz * 3 * Gd * 4);
  auto* solvB = (unsigned short*)alloc((size_t)Bsz * SOLV * 2);
  auto* s1B   = (unsigned short*)alloc((size_t)Bsz * Gd * 2);
  float* combF = (float*)alloc((size_t)Bsz * 1024 * 4);
  auto* combB  = (unsigned short*)alloc((size_t)Bsz * 1024 * 2);
  float* hidF  = (float*)alloc((size_t)Bsz * 128 * 4);
  float* hvpF  = (float*)he1B;  // alias, free after et-GEMM

  auto blocks = [](size_t n) { return dim3((unsigned)((n + 255) / 256)); };
  auto gemm = [&](const unsigned short* A, const unsigned short* W, const float* bi,
                  float* C, unsigned short* Cb, int M, int N, int K, int ldc, int act) {
    gemm_bf16_wmma<<<dim3(M / 64, N / 128), 256, 0, stream>>>(A, W, bi, C, Cb, M, N, K, ldc, act);
  };
  auto castW = [&](const float* s, unsigned short* d, size_t n) {
    cast_bf<<<blocks(n), 256, 0, stream>>>(s, d, (int)n);
  };

  // ---- weight prep --------------------------------------------------------
  pack_pad<<<blocks((size_t)Gd * 64), 256, 0, stream>>>(W_pn, WpnB, Gd, NFd, 64);
  pack_pad<<<blocks((size_t)Gd * 64), 256, 0, stream>>>(W_pe1, Wpe1B, Gd, NFd + EFd, 64);
  castW(W_et, WetB, (size_t)Gd * Gd);
  castW(Wih1, Wih1B, (size_t)3 * Gd * Gd);   castW(Whh1, Whh1B, (size_t)3 * Gd * Gd);
  castW(W_pnode, WpnodeB, (size_t)Gd * Gd);
  castW(Wih2, Wih2B, (size_t)3 * Gd * Gd);   castW(Whh2, Whh2B, (size_t)3 * Gd * Gd);
  castW(W_proj, WprojB, (size_t)Td * Gd * Gd);
  castW(Wih_r, WihrB, (size_t)Td * 3 * Gd * Gd);
  castW(Whh_r, WhhrB, (size_t)Td * 3 * Gd * Gd);
  castW(Ws1, Ws1B, (size_t)Gd * SOLV);       castW(Ws2, Ws2B, (size_t)Gd * Gd);
  castW(Wp1, Wp1B, (size_t)128 * SOLV);

  // ---- input prep ---------------------------------------------------------
  pack_pad<<<blocks((size_t)Vn * 64), 256, 0, stream>>>(node_feats, XnB, Vn, NFd, 64);
  pack_edge_in<<<En, 64, 0, stream>>>(node_feats, edge_feats, srcI, EinB);
  pack_solvent<<<blocks((size_t)Bsz * SOLV), 256, 0, stream>>>(fps, solvB, Bsz * SOLV);

  // ---- node / edge projections (WMMA) -------------------------------------
  gemm(XnB, WpnB, b_pn, hvF, hvB, Vn, Gd, 64, Gd, 1);        // hv_new = lrelu
  gemm(EinB, Wpe1B, b_pe1, R1, he1B, En, Gd, 64, Gd, 1);     // he1 = lrelu

  // ---- attention round 1 --------------------------------------------------
  concat_dot<<<En / 8, 256, 0, stream>>>(hvF, dstI, R1, nullptr, W_pe2, b_pe2, lgBuf, En, 0);
  fill_f32<<<blocks(Vn), 256, 0, stream>>>(mBuf, -1e30f, Vn);
  fill_f32<<<blocks(Vn), 256, 0, stream>>>(sBuf, 0.f, Vn);
  seg_max<<<blocks(En), 256, 0, stream>>>(lgBuf, dstI, mBuf, En);
  seg_expsum<<<blocks(En), 256, 0, stream>>>(lgBuf, dstI, mBuf, exBuf, sBuf, En);
  gemm(he1B, WetB, b_et, R2, nullptr, En, Gd, Gd, Gd, 0);    // he1 @ W_et + b
  fill_f32<<<blocks((size_t)Vn * Gd), 256, 0, stream>>>(ctxF, 0.f, Vn * Gd);
  scatter_msg<<<En, Gd, 0, stream>>>(R2, exBuf, sBuf, dstI, ctxF);
  elu_cast_bf<<<blocks((size_t)Vn * Gd), 256, 0, stream>>>(ctxF, ctxB, Vn * Gd);

  // ---- GRU 1 --------------------------------------------------------------
  gemm(ctxB, Wih1B, bih1, R1, nullptr, Vn, 3 * Gd, Gd, 3 * Gd, 0);
  gemm(hvB,  Whh1B, bhh1, R2, nullptr, Vn, 3 * Gd, Gd, 3 * Gd, 0);
  gru_combine<<<blocks((size_t)Vn * Gd), 256, 0, stream>>>(R1, R2, hvF, hF, hB, Vn, 1);

  // ---- attention round 2 --------------------------------------------------
  concat_dot<<<En / 8, 256, 0, stream>>>(hF, dstI, hF, srcI, W_pe, b_pe, lgBuf, En, 0);
  fill_f32<<<blocks(Vn), 256, 0, stream>>>(mBuf, -1e30f, Vn);
  fill_f32<<<blocks(Vn), 256, 0, stream>>>(sBuf, 0.f, Vn);
  seg_max<<<blocks(En), 256, 0, stream>>>(lgBuf, dstI, mBuf, En);
  seg_expsum<<<blocks(En), 256, 0, stream>>>(lgBuf, dstI, mBuf, exBuf, sBuf, En);
  gemm(hB, WpnodeB, b_pnode, hvpF, nullptr, Vn, Gd, Gd, Gd, 0);   // hv = h @ W_pnode
  fill_f32<<<blocks((size_t)Vn * Gd), 256, 0, stream>>>(ctxF, 0.f, Vn * Gd);
  scatter_src<<<En, Gd, 0, stream>>>(hvpF, exBuf, sBuf, srcI, dstI, ctxF);
  elu_cast_bf<<<blocks((size_t)Vn * Gd), 256, 0, stream>>>(ctxF, ctxB, Vn * Gd);

  // ---- GRU 2  (h2 written into hv buffers) --------------------------------
  gemm(ctxB, Wih2B, bih2, R1, nullptr, Vn, 3 * Gd, Gd, 3 * Gd, 0);
  gemm(hB,   Whh2B, bhh2, R2, nullptr, Vn, 3 * Gd, Gd, 3 * Gd, 0);
  gru_combine<<<blocks((size_t)Vn * Gd), 256, 0, stream>>>(R1, R2, hF, hvF, hvB, Vn, 1);

  // ---- readout ------------------------------------------------------------
  graph_pool<<<Bsz, Gd, 0, stream>>>(hvF, nullptr, gfeatF);
  for (int t = 0; t < Td; ++t) {
    cast_bf<<<blocks((size_t)Bsz * Gd), 256, 0, stream>>>(gfeatF, gfeatB, Bsz * Gd);
    concat_dot<<<Vn / 8, 256, 0, stream>>>(gfeatF, n2g, hvF, nullptr,
                                           W_logit + t * 2 * Gd, b_logit + t, lgBuf, Vn, 1);
    graph_softmax<<<Bsz, 64, 0, stream>>>(lgBuf, exBuf);
    gemm(hvB, WprojB + (size_t)t * Gd * Gd, b_proj + t * Gd, hvpF, nullptr, Vn, Gd, Gd, Gd, 0);
    graph_pool<<<Bsz, Gd, 0, stream>>>(hvpF, exBuf, grprF);
    elu_cast_bf<<<blocks((size_t)Bsz * Gd), 256, 0, stream>>>(grprF, grprB, Bsz * Gd);
    gemm(grprB,  WihrB + (size_t)t * 3 * Gd * Gd, bih_r + t * 3 * Gd, giR, nullptr, Bsz, 3 * Gd, Gd, 3 * Gd, 0);
    gemm(gfeatB, WhhrB + (size_t)t * 3 * Gd * Gd, bhh_r + t * 3 * Gd, ghR, nullptr, Bsz, 3 * Gd, Gd, 3 * Gd, 0);
    gru_combine<<<blocks((size_t)Bsz * Gd), 256, 0, stream>>>(giR, ghR, gfeatF, gfeatF, nullptr, Bsz, 1);
  }
  copy_cols<<<Bsz, Gd, 0, stream>>>(gfeatF, combF, Gd, 1024);

  // ---- solvent MLP (WMMA) -------------------------------------------------
  gemm(solvB, Ws1B, bs1, nullptr, s1B, Bsz, Gd, SOLV, Gd, 2);       // relu
  gemm(s1B, Ws2B, bs2, combF + Gd, nullptr, Bsz, Gd, Gd, 1024, 0);  // -> cols 256..511

  // ---- fused smiles conv-attention ----------------------------------------
  smiles_conv<<<dim3(Bsz, Gd / 8), 256, 0, stream>>>(fps, Wcf, bcf, Wca, bca, combF, 1024);

  // ---- prediction head ----------------------------------------------------
  cast_bf<<<blocks((size_t)Bsz * 1024), 256, 0, stream>>>(combF, combB, Bsz * 1024);
  gemm(combB, Wp1B, bp1, hidF, nullptr, Bsz, 128, 1024, 128, 2);    // relu
  final_dot<<<Bsz / 8, 256, 0, stream>>>(hidF, Wp2, bp2, (float*)d_out);
}
